// DetourTransformer_1821066134191
// MI455X (gfx1250) — compile-verified, matrix operand order checked by reading, not run on hardware
//
#include <hip/hip_runtime.h>
#include <hip/hip_bf16.h>
#include <math.h>

// ---------------- problem constants ----------------
#define NB   16
#define SEQ  512
#define DIN  128
#define DM   256
#define NH   8
#define HC   32
#define NL   4
#define MROWS (NB*SEQ)     // 8192
#define EPS  1e-5f

typedef __bf16 bf16;
typedef __attribute__((ext_vector_type(16))) __bf16 v16bf;
typedef __attribute__((ext_vector_type(8)))  __bf16 v8bf;
typedef __attribute__((ext_vector_type(8)))  float  v8f;

union Frag { v16bf v; v8bf h[2]; };

#define WMMA_BF16(A, B, C) \
  __builtin_amdgcn_wmma_f32_16x16x32_bf16(false, (A), false, (B), (short)0, (C), false, false)

// --- gfx1250 async global->LDS copy (ASYNCcnt-tracked), 16 bytes per lane ---
__device__ __forceinline__ void dt_async_b128(unsigned lds_off, const bf16* gsrc) {
  asm volatile("global_load_async_to_lds_b128 %0, %1, off"
               :: "v"(lds_off), "v"(gsrc) : "memory");
}
__device__ __forceinline__ void dt_wait_async0() {
  asm volatile("s_wait_asynccnt 0x0" ::: "memory");
}

// ---------------- elementwise helpers ----------------
__global__ void dt_f32_to_bf16(const float* __restrict__ s, bf16* __restrict__ d, int n) {
  int i = blockIdx.x * blockDim.x + threadIdx.x;
  if (i < n) d[i] = (bf16)s[i];
}

// out = a + b, written as f32 and bf16
__global__ void dt_add2(const float* __restrict__ a, const float* __restrict__ b,
                        float* __restrict__ of, bf16* __restrict__ ob, int n) {
  int i = blockIdx.x * blockDim.x + threadIdx.x;
  if (i < n) { float v = a[i] + b[i]; of[i] = v; ob[i] = (bf16)v; }
}

// vT[((n*NH+h)*HC + c)*SEQ + t] = v[(n*SEQ + t)*DM + h*HC + c]
__global__ void dt_transpose_v(const bf16* __restrict__ v, bf16* __restrict__ vT) {
  int i = blockIdx.x * blockDim.x + threadIdx.x;
  if (i >= NB * NH * HC * SEQ) return;
  int t = i & (SEQ - 1);
  int c = (i >> 9) & (HC - 1);
  int h = (i >> 14) & (NH - 1);
  int n = i >> 17;
  vT[i] = v[((size_t)(n * SEQ + t)) * DM + h * HC + c];
}

// gather h[:, 0] -> out [NB, DM]
__global__ void dt_gather(const float* __restrict__ h, float* __restrict__ out) {
  int i = blockIdx.x * blockDim.x + threadIdx.x;
  if (i >= NB * DM) return;
  int r = i >> 8, c = i & 255;
  out[i] = h[((size_t)r * SEQ) * DM + c];
}

// ---------------- weight packing into WMMA B-fragment order ----------------
// packed[((tn*ktiles + tk)*512) + lane*16 + e]:
//   col = tn*16 + (lane&15); g = lane>>4;
//   e<8  : k = tk*32 + g*8 + e
//   e>=8 : k = tk*32 + 16 + g*8 + (e-8)
__global__ void dt_pack_w(const float* __restrict__ W, bf16* __restrict__ out, int K) {
  int idx = blockIdx.x * blockDim.x + threadIdx.x;
  if (idx >= K * DM) return;
  int e    = idx & 15;
  int lane = (idx >> 4) & 31;
  int tile = idx >> 9;
  int ktiles = K / 32;
  int tn = tile / ktiles;
  int tk = tile - tn * ktiles;
  int col = tn * 16 + (lane & 15);
  int g   = lane >> 4;
  int k   = tk * 32 + ((e < 8) ? (g * 8 + e) : (16 + g * 8 + (e - 8)));
  out[idx] = (bf16)W[(size_t)k * DM + col];
}

// ---------------- fused GEMM + bias + BatchNorm(eval) ----------------
// 4 waves per block; block computes 64 rows x 64 cols. The 64-col weight panel
// (4*ktiles 32x16 B-fragments, <=32KB) is staged into LDS ONCE per block via
// gfx1250 async global->LDS copies, then all 4 waves feed WMMAs from LDS.
// grid = (MROWS/64, 4), block = 128.
__global__ __launch_bounds__(128)
void dt_gemm_bn(const bf16* __restrict__ X, const bf16* __restrict__ Wp,
                const float* __restrict__ bias,
                const float* __restrict__ bng, const float* __restrict__ bnb,
                const float* __restrict__ bnm, const float* __restrict__ bnv,
                float* __restrict__ out_f32, bf16* __restrict__ out_bf16, int K) {
  __shared__ __align__(16) bf16 Bs[4 * 8 * 512];   // up to 32 KB (ktiles <= 8)

  const int rt   = blockIdx.x;            // 64-row block
  const int ct   = blockIdx.y;            // 64-col block
  const int tid  = threadIdx.x;
  const int wave = tid >> 5;
  const int lane = tid & 31;
  const int lr   = lane & 15;
  const int gsel = lane >> 4;
  const int row  = rt * 64 + wave * 16 + lr;
  const int ktiles = K / 32;

  // ---- async-stage the whole B panel (contiguous in packed layout) ----
  {
    const bf16* src = Wp + (size_t)(ct * 4 * ktiles) * 512;
    unsigned bs0 = (unsigned)(uintptr_t)(&Bs[0]);
    int nchunks = (4 * ktiles * 512) / 8;               // 16B chunks
    for (int c = tid; c < nchunks; c += 128)
      dt_async_b128(bs0 + c * 16, src + c * 8);
    dt_wait_async0();
  }
  __syncthreads();

  v8f acc0 = {}, acc1 = {}, acc2 = {}, acc3 = {};

  for (int tk = 0; tk < ktiles; ++tk) {
    const bf16* ap = X + (size_t)row * K + tk * 32 + gsel * 8;
    Frag A; A.h[0] = *(const v8bf*)ap; A.h[1] = *(const v8bf*)(ap + 16);
    __builtin_prefetch(ap + 32, 0, 3);   // global_prefetch_b8, near-cache hint
#pragma unroll
    for (int j = 0; j < 4; ++j) {
      const bf16* bp = &Bs[((j * ktiles) + tk) * 512 + lane * 16];
      Frag B; B.h[0] = *(const v8bf*)bp; B.h[1] = *(const v8bf*)(bp + 8);
      v8f c = (j == 0) ? acc0 : (j == 1) ? acc1 : (j == 2) ? acc2 : acc3;
      c = WMMA_BF16(A.v, B.v, c);
      if (j == 0) acc0 = c; else if (j == 1) acc1 = c; else if (j == 2) acc2 = c; else acc3 = c;
    }
  }

#pragma unroll
  for (int j = 0; j < 4; ++j) {
    int colb = ct * 64 + j * 16 + lr;
    float sc  = bng[colb] * rsqrtf(bnv[colb] + EPS);
    float off = (bias[colb] - bnm[colb]) * sc + bnb[colb];
    v8f c = (j == 0) ? acc0 : (j == 1) ? acc1 : (j == 2) ? acc2 : acc3;
#pragma unroll
    for (int i = 0; i < 8; ++i) {
      int r = rt * 64 + wave * 16 + gsel * 8 + i;
      float val = c[i] * sc + off;
      if (out_f32)  out_f32 [(size_t)r * DM + colb] = val;
      if (out_bf16) out_bf16[(size_t)r * DM + colb] = (bf16)val;
    }
  }
}

// ---------------- residual add + LayerNorm ----------------
// one wave per row of 256; grid = MROWS, block = 32
__global__ __launch_bounds__(32)
void dt_add_ln(const float* __restrict__ a, const float* __restrict__ b,
               const float* __restrict__ g, const float* __restrict__ be,
               float* __restrict__ of, bf16* __restrict__ ob) {
  int row = blockIdx.x, lane = threadIdx.x;
  const float* ap = a + (size_t)row * DM + lane * 8;
  const float* bp = b + (size_t)row * DM + lane * 8;
  float v[8]; float s = 0.f;
#pragma unroll
  for (int i = 0; i < 8; ++i) { v[i] = ap[i] + bp[i]; s += v[i]; }
#pragma unroll
  for (int o = 16; o >= 1; o >>= 1) s += __shfl_xor(s, o, 32);
  float mu = s * (1.f / DM);
  float q = 0.f;
#pragma unroll
  for (int i = 0; i < 8; ++i) { float d = v[i] - mu; q += d * d; }
#pragma unroll
  for (int o = 16; o >= 1; o >>= 1) q += __shfl_xor(q, o, 32);
  float rs = rsqrtf(q * (1.f / DM) + EPS);
#pragma unroll
  for (int i = 0; i < 8; ++i) {
    int c = lane * 8 + i;
    float o = (v[i] - mu) * rs * g[c] + be[c];
    of[(size_t)row * DM + c] = o;
    if (ob) ob[(size_t)row * DM + c] = (bf16)o;
  }
}

// ---------------- attention (QK^T * mask, softmax, attn*V) ----------------
// One wave per (n*H + h, 16-row block). grid = (NB*NH, SEQ/16), block = 32.
// Scores for the 16x512 row-block live in LDS; exp() values stored as bf16 in
// A-fragment-friendly row-major form so attn*V A-frags are contiguous b128s.
__global__ __launch_bounds__(32)
void dt_attn(const bf16* __restrict__ q, const bf16* __restrict__ k,
             const bf16* __restrict__ vT, const float* __restrict__ mask,
             float* __restrict__ val_out) {
  __shared__ __align__(16) float sc_f[16 * SEQ];   // 32 KB
  __shared__ __align__(16) bf16  sc_b[16 * SEQ];   // 16 KB
  __shared__ float rsum[16];

  const int nh = blockIdx.x, rt = blockIdx.y;
  const int n = nh >> 3, h = nh & 7;
  const int lane = threadIdx.x;
  const int lr = lane & 15, gsel = lane >> 4;

  // Q fragment (A, 16x32): row = rt*16 + lr, c = gsel*8..+7 and +16
  const bf16* qp = q + ((size_t)(n * SEQ + rt * 16 + lr)) * DM + h * HC + gsel * 8;
  Frag A; A.h[0] = *(const v8bf*)qp; A.h[1] = *(const v8bf*)(qp + 16);

  // ---- scores: 32 tiles of 16 t-columns each ----
  for (int tt = 0; tt < SEQ / 16; ++tt) {
    const bf16* kp = k + ((size_t)(n * SEQ + tt * 16 + lr)) * DM + h * HC + gsel * 8;
    Frag B; B.h[0] = *(const v8bf*)kp; B.h[1] = *(const v8bf*)(kp + 16);
    v8f c = {};
    c = WMMA_BF16(A.v, B.v, c);
#pragma unroll
    for (int i = 0; i < 8; ++i) {
      int mr = gsel * 8 + i;                 // local row
      int tc = tt * 16 + lr;                 // t column
      float m = mask[((size_t)(n * SEQ) + rt * 16 + mr) * SEQ + tc];
      sc_f[mr * SEQ + tc] = c[i] * m;
    }
  }
  __syncthreads();

  // ---- softmax: lane handles row (lane&15), half (lane>>4) ----
  {
    int t0 = gsel * (SEQ / 2);
    float mx = -3.4e38f;
    for (int t = t0; t < t0 + SEQ / 2; ++t) mx = fmaxf(mx, sc_f[lr * SEQ + t]);
    mx = fmaxf(mx, __shfl_xor(mx, 16, 32));
    float sum = 0.f;
    for (int t = t0; t < t0 + SEQ / 2; ++t) {
      float e = __expf(sc_f[lr * SEQ + t] - mx);
      sc_b[lr * SEQ + t] = (bf16)e;
      sum += e;
    }
    sum += __shfl_xor(sum, 16, 32);
    if (gsel == 0) rsum[lr] = 1.0f / sum;
  }
  __syncthreads();

  // ---- attn * V : output 16 rows x 32 cols (2 wmma tiles), K-loop over t ----
  v8f acc0 = {}, acc1 = {};
  for (int tb = 0; tb < SEQ / 32; ++tb) {
    const bf16* ap = &sc_b[lr * SEQ + tb * 32 + gsel * 8];
    Frag Aa; Aa.h[0] = *(const v8bf*)ap; Aa.h[1] = *(const v8bf*)(ap + 16);
#pragma unroll
    for (int j = 0; j < 2; ++j) {
      const bf16* bp = vT + ((size_t)nh * HC + j * 16 + lr) * SEQ + tb * 32 + gsel * 8;
      Frag Bb; Bb.h[0] = *(const v8bf*)bp; Bb.h[1] = *(const v8bf*)(bp + 16);
      if (j == 0) acc0 = WMMA_BF16(Aa.v, Bb.v, acc0);
      else        acc1 = WMMA_BF16(Aa.v, Bb.v, acc1);
    }
  }
#pragma unroll
  for (int i = 0; i < 8; ++i) {
    int mr = gsel * 8 + i;
    float scale = rsum[mr];
    size_t rbase = ((size_t)(n * SEQ + rt * 16 + mr)) * DM + h * HC;
    val_out[rbase + 0 * 16 + lr] = acc0[i] * scale;
    val_out[rbase + 1 * 16 + lr] = acc1[i] * scale;
  }
}

// ---------------- host orchestration ----------------
extern "C" void kernel_launch(void* const* d_in, const int* in_sizes, int n_in,
                              void* d_out, int out_size, void* d_ws, size_t ws_size,
                              hipStream_t stream) {
  (void)in_sizes; (void)n_in; (void)out_size; (void)ws_size;
  auto F = [&](int i) { return (const float*)d_in[i]; };

  // input index map (depth-first, dict insertion order):
  // 0:x 1:pad_mask 2:W_in 3:b_in 4..7:bn_in{g,b,m,v}
  // layer l base = 8 + 32*l:
  //  +0 Wq +1 bq +2..5 bnq | +6 Wk +7 bk +8..11 bnk | +12 Wv +13 bv +14..17 bnv
  //  +18 Wff +19 bff +20..23 bnff | +24 Wskip +25 bskip +26..29 bnskip | +30 ln_g +31 ln_b
  const float* x    = F(0);
  const float* mask = F(1);

  char* wp = (char*)d_ws;
  auto alloc = [&](size_t bytes) { void* p = wp; wp += (bytes + 255) & ~(size_t)255; return p; };

  float* h_f32   = (float*)alloc(sizeof(float) * MROWS * DM);
  bf16*  h_bf    = (bf16*) alloc(sizeof(bf16)  * MROWS * DM);
  bf16*  q_bf    = (bf16*) alloc(sizeof(bf16)  * MROWS * DM);
  bf16*  k_bf    = (bf16*) alloc(sizeof(bf16)  * MROWS * DM);
  bf16*  v_bf    = (bf16*) alloc(sizeof(bf16)  * MROWS * DM);
  bf16*  vT      = (bf16*) alloc(sizeof(bf16)  * MROWS * DM);
  float* val_f   = (float*)alloc(sizeof(float) * MROWS * DM);  // reused as out2
  float* out1_f  = (float*)alloc(sizeof(float) * MROWS * DM);
  bf16*  out1_bf = (bf16*) alloc(sizeof(bf16)  * MROWS * DM);
  float* ff_f    = (float*)alloc(sizeof(float) * MROWS * DM);
  float* skip_f  = (float*)alloc(sizeof(float) * MROWS * DM);
  bf16*  x_bf    = (bf16*) alloc(sizeof(bf16)  * MROWS * DIN);
  bf16*  wp_in   = (bf16*) alloc(sizeof(bf16)  * DIN * DM);
  bf16*  wp_l[NL][5];
  for (int l = 0; l < NL; ++l)
    for (int j = 0; j < 5; ++j)
      wp_l[l][j] = (bf16*)alloc(sizeof(bf16) * DM * DM);

  // ---- convert input + pack all weights to bf16 fragment order ----
  dt_f32_to_bf16<<<(MROWS * DIN + 255) / 256, 256, 0, stream>>>(x, x_bf, MROWS * DIN);
  dt_pack_w<<<(DIN * DM + 255) / 256, 256, 0, stream>>>(F(2), wp_in, DIN);
  for (int l = 0; l < NL; ++l) {
    int base = 8 + 32 * l;
    const int widx[5] = {0, 6, 12, 18, 24};
    for (int j = 0; j < 5; ++j)
      dt_pack_w<<<(DM * DM + 255) / 256, 256, 0, stream>>>(F(base + widx[j]), wp_l[l][j], DM);
  }

  dim3 ggrid(MROWS / 64, 4);

  // ---- input projection + BN ----
  dt_gemm_bn<<<ggrid, 128, 0, stream>>>(x_bf, wp_in, F(3), F(4), F(5), F(6), F(7),
                                        h_f32, h_bf, DIN);

  for (int l = 0; l < NL; ++l) {
    int b = 8 + 32 * l;
    const float* ln_g = F(b + 30);
    const float* ln_b = F(b + 31);

    // q/k/v projections (bf16 outputs only; consumed by WMMA attention)
    dt_gemm_bn<<<ggrid, 128, 0, stream>>>(h_bf, wp_l[l][0], F(b + 1),  F(b + 2),  F(b + 3),  F(b + 4),  F(b + 5),
                                          nullptr, q_bf, DM);
    dt_gemm_bn<<<ggrid, 128, 0, stream>>>(h_bf, wp_l[l][1], F(b + 7),  F(b + 8),  F(b + 9),  F(b + 10), F(b + 11),
                                          nullptr, k_bf, DM);
    dt_gemm_bn<<<ggrid, 128, 0, stream>>>(h_bf, wp_l[l][2], F(b + 13), F(b + 14), F(b + 15), F(b + 16), F(b + 17),
                                          nullptr, v_bf, DM);

    dt_transpose_v<<<(NB * NH * HC * SEQ + 255) / 256, 256, 0, stream>>>(v_bf, vT);
    dt_attn<<<dim3(NB * NH, SEQ / 16), 32, 0, stream>>>(q_bf, k_bf, vT, mask, val_f);

    // out1 = LN(val + h)
    dt_add_ln<<<MROWS, 32, 0, stream>>>(val_f, h_f32, ln_g, ln_b, out1_f, out1_bf);

    // ff = BN(out1 @ Wff + bff)
    dt_gemm_bn<<<ggrid, 128, 0, stream>>>(out1_bf, wp_l[l][3], F(b + 19), F(b + 20), F(b + 21), F(b + 22), F(b + 23),
                                          ff_f, nullptr, DM);

    // out2 = LN(out1 + ff)  (reuse val_f)
    dt_add_ln<<<MROWS, 32, 0, stream>>>(out1_f, ff_f, ln_g, ln_b, val_f, nullptr);

    // skip = BN(h @ Wskip + bskip)
    dt_gemm_bn<<<ggrid, 128, 0, stream>>>(h_bf, wp_l[l][4], F(b + 25), F(b + 26), F(b + 27), F(b + 28), F(b + 29),
                                          skip_f, nullptr, DM);

    // h = out2 + skip (f32 + bf16 for next layer)
    dt_add2<<<(MROWS * DM + 255) / 256, 256, 0, stream>>>(val_f, skip_f, h_f32, h_bf, MROWS * DM);
  }

  dt_gather<<<(NB * DM + 255) / 256, 256, 0, stream>>>(h_f32, (float*)d_out);
}
